// GroupedQueryAttention_515396075669
// MI455X (gfx1250) — matrix operand order repaired and checked by reading, not compiled, not run on hardware
//
#include <hip/hip_runtime.h>

#define DIMN 2048
#define SEQ  2048
#define BATCH 2
#define NH   16
#define NKV  4
#define HD   128
#define MTOT (BATCH*SEQ)   /* 4096 */
#define DKV  (NKV*HD)      /* 512  */

typedef __attribute__((ext_vector_type(16))) __bf16 v16bf;
typedef __attribute__((ext_vector_type(8)))  float  v8f;
typedef __attribute__((ext_vector_type(4)))  int    v4i;

struct __align__(8) bf16x4 { __bf16 d[4]; };

union FragU { uint4 u[2]; v16bf v; };

static __device__ inline v16bf ld_frag2(const __bf16* p0, const __bf16* p1) {
  FragU f;
  f.u[0] = *reinterpret_cast<const uint4*>(p0);
  f.u[1] = *reinterpret_cast<const uint4*>(p1);
  return f.v;
}

static __device__ inline v8f wmma_bf16f32(v16bf a, v16bf b, v8f c) {
  // (neg_a, A, neg_b, B, c_mod, C, reuse_a, reuse_b)
  return __builtin_amdgcn_wmma_f32_16x16x32_bf16(false, a, false, b, (short)0, c, false, false);
}

// ---------------------------------------------------------------------------
// Async Global -> LDS copy (CDNA5 GLOBAL_LOAD_ASYNC_TO_LDS_B128, ASYNCcnt),
// with a synchronous fallback if the builtin is unavailable.
// Probe result (round 2 diagnostic): param0 is AS1 `int4*`, so cast to v4i*.
// ---------------------------------------------------------------------------
#if defined(__has_builtin)
#if __has_builtin(__builtin_amdgcn_global_load_async_to_lds_b128)
#define HAVE_ASYNC_LDS 1
#endif
#endif

#ifdef HAVE_ASYNC_LDS
#define AS_GLOBAL __attribute__((address_space(1)))
#define AS_LDS    __attribute__((address_space(3)))
static __device__ inline void async_copy16(__bf16* lds_dst, const __bf16* gsrc) {
  __builtin_amdgcn_global_load_async_to_lds_b128(
      (AS_GLOBAL v4i*)(uintptr_t)gsrc,   // flat global addr == AS1 addr
      (AS_LDS v4i*)lds_dst,              // addrspacecast flat -> lds
      0, 0);
}
static __device__ inline void wait_async0() {
#if __has_builtin(__builtin_amdgcn_s_wait_asynccnt)
  __builtin_amdgcn_s_wait_asynccnt(0);
#else
  asm volatile("s_wait_asynccnt 0x0" ::: "memory");
#endif
}
#else
static __device__ inline void async_copy16(__bf16* lds_dst, const __bf16* gsrc) {
  *reinterpret_cast<uint4*>(lds_dst) = *reinterpret_cast<const uint4*>(gsrc);
}
static __device__ inline void wait_async0() {}
#endif

// ---------------------------------------------------------------------------
// f32 -> bf16 convert (vectorized x4)
// ---------------------------------------------------------------------------
__global__ void gqa_f32_to_bf16(const float* __restrict__ in, __bf16* __restrict__ out, int total4) {
  const int t = blockIdx.x * blockDim.x + threadIdx.x;
  if (t >= total4) return;
  float4 v = *reinterpret_cast<const float4*>(in + (size_t)t * 4);
  bf16x4 o;
  o.d[0] = (__bf16)v.x; o.d[1] = (__bf16)v.y; o.d[2] = (__bf16)v.z; o.d[3] = (__bf16)v.w;
  *reinterpret_cast<bf16x4*>(out + (size_t)t * 4) = o;
}

// f32 [K][N] -> bf16 transposed [N][K]  (one-time; keeps hot GEMM loop copy-only)
__global__ void gqa_f32_to_bf16_t(const float* __restrict__ in, __bf16* __restrict__ out,
                                  int N, int K, int total4) {
  const int t = blockIdx.x * blockDim.x + threadIdx.x;
  if (t >= total4) return;
  const size_t i4 = (size_t)t * 4;
  const int n = (int)(i4 % N);
  const int k = (int)(i4 / N);
  float4 v = *reinterpret_cast<const float4*>(in + i4);
  out[(size_t)(n + 0) * K + k] = (__bf16)v.x;
  out[(size_t)(n + 1) * K + k] = (__bf16)v.y;
  out[(size_t)(n + 2) * K + k] = (__bf16)v.z;
  out[(size_t)(n + 3) * K + k] = (__bf16)v.w;
}

// ---------------------------------------------------------------------------
// bf16 GEMM: C[M,N] = A[M,K] x B[K,N], with B pre-transposed as Bt[N][K].
// 128x128 block, BK=32, 256 threads = 8 waves (each 64x32 = 4x2 WMMA tiles).
// Double-buffered LDS, async global->LDS staging overlapped with WMMA.
// ---------------------------------------------------------------------------
__global__ __launch_bounds__(256) void gqa_gemm_bf16f32(const __bf16* __restrict__ A,
                                                        const __bf16* __restrict__ Bt,
                                                        float* __restrict__ C,
                                                        int N, int K) {
  __shared__ __bf16 sA[2][128 * 32];    // [m][k]
  __shared__ __bf16 sBt[2][128 * 32];   // [n][k]
  const int tid  = threadIdx.x;
  const int lane = tid & 31, wave = tid >> 5;
  const int lg = lane >> 4, ll = lane & 15;
  const int wm = wave >> 2, wn = wave & 3;
  const int m0 = blockIdx.x * 128;
  const int n0 = blockIdx.y * 128;

  const int sr = tid >> 1;              // staging row 0..127
  const int sc = (tid & 1) * 16;        // staging col (elements)

  const v8f vzero = {0.f, 0.f, 0.f, 0.f, 0.f, 0.f, 0.f, 0.f};
  v8f acc[4][2];
#pragma unroll
  for (int mi = 0; mi < 4; ++mi)
#pragma unroll
    for (int ni = 0; ni < 2; ++ni) acc[mi][ni] = vzero;

  const int ksteps = K >> 5;

  // prologue: stage tile 0
  {
    const __bf16* ga = A  + (size_t)(m0 + sr) * K + sc;
    const __bf16* gb = Bt + (size_t)(n0 + sr) * K + sc;
    async_copy16(sA[0]  + sr * 32 + sc,     ga);
    async_copy16(sA[0]  + sr * 32 + sc + 8, ga + 8);
    async_copy16(sBt[0] + sr * 32 + sc,     gb);
    async_copy16(sBt[0] + sr * 32 + sc + 8, gb + 8);
  }
  wait_async0();
  __syncthreads();

  for (int kb = 0; kb < ksteps; ++kb) {
    const int cur = kb & 1;
    if (kb + 1 < ksteps) {               // stage next tile while computing
      const int nxt = cur ^ 1;
      const __bf16* ga = A  + (size_t)(m0 + sr) * K + (kb + 1) * 32 + sc;
      const __bf16* gb = Bt + (size_t)(n0 + sr) * K + (kb + 1) * 32 + sc;
      async_copy16(sA[nxt]  + sr * 32 + sc,     ga);
      async_copy16(sA[nxt]  + sr * 32 + sc + 8, ga + 8);
      async_copy16(sBt[nxt] + sr * 32 + sc,     gb);
      async_copy16(sBt[nxt] + sr * 32 + sc + 8, gb + 8);
    }
    // compute on current buffer
    v16bf bfr[2];
#pragma unroll
    for (int ni = 0; ni < 2; ++ni) {
      const __bf16* bp = sBt[cur] + (wn * 32 + ni * 16 + ll) * 32 + 16 * lg;
      bfr[ni] = ld_frag2(bp, bp + 8);
    }
#pragma unroll
    for (int mi = 0; mi < 4; ++mi) {
      const __bf16* ap = sA[cur] + (wm * 64 + mi * 16 + ll) * 32 + 16 * lg;
      v16bf af = ld_frag2(ap, ap + 8);
#pragma unroll
      for (int ni = 0; ni < 2; ++ni)
        acc[mi][ni] = wmma_bf16f32(af, bfr[ni], acc[mi][ni]);
    }
    wait_async0();
    __syncthreads();
  }
  // epilogue: C/D layout row = r + 8*lg, col = ll
#pragma unroll
  for (int mi = 0; mi < 4; ++mi)
#pragma unroll
    for (int ni = 0; ni < 2; ++ni)
#pragma unroll
      for (int r = 0; r < 8; ++r) {
        const int row = m0 + wm * 64 + mi * 16 + r + 8 * lg;
        const int col = n0 + wn * 32 + ni * 16 + ll;
        C[(size_t)row * N + col] = acc[mi][ni][r];
      }
}

// ---------------------------------------------------------------------------
// RMSNorm + RoPE: one wave32 per (token, head) row of 128; output bf16
// rearranged [b, head, s, d].
// ---------------------------------------------------------------------------
__global__ __launch_bounds__(128) void gqa_rmsnorm_rope(const float* __restrict__ in,
                                                        const float* __restrict__ w,
                                                        __bf16* __restrict__ out,
                                                        int H, int nrows) {
  const int warp = blockIdx.x * 4 + (threadIdx.x >> 5);
  if (warp >= nrows) return;
  const int lane = threadIdx.x & 31;
  const int h = warp % H;
  const int s = (warp / H) % SEQ;
  const int b = warp / (H * SEQ);

  float4 x = *reinterpret_cast<const float4*>(in + (size_t)warp * HD + lane * 4);
  float ss = x.x * x.x + x.y * x.y + x.z * x.z + x.w * x.w;
#pragma unroll
  for (int m = 16; m >= 1; m >>= 1) ss += __shfl_xor(ss, m, 32);
  const float inv = rsqrtf(ss * (1.0f / 128.0f) + 1e-6f);
  float4 wv = *reinterpret_cast<const float4*>(w + lane * 4);
  float xn[4] = {x.x * inv * wv.x, x.y * inv * wv.y, x.z * inv * wv.z, x.w * inv * wv.w};

  bf16x4 ov;
#pragma unroll
  for (int i = 0; i < 4; ++i) {
    const int d = lane * 4 + i;
    const int j = d & 63;
    const float ang = (float)s * __expf(-(float)j * 0.14391156831f); // ln(1e4)/64
    const float c = __cosf(ang), si = __sinf(ang);
    const float other = __shfl_xor(xn[i], 16, 32); // partner dim d^64 lives in lane^16
    const float rot = (d < 64) ? -other : other;
    ov.d[i] = (__bf16)(xn[i] * c + rot * si);
  }
  *reinterpret_cast<bf16x4*>(out + (((size_t)(b * H + h) * SEQ + s) * HD + lane * 4)) = ov;
}

// ---------------------------------------------------------------------------
// V rearrange + convert + transpose: f32 [b,s,kv,d] -> bf16 [b,kv,d,s] (V^T)
// so the attention hot loop stages V with straight 16B copies.
// ---------------------------------------------------------------------------
__global__ void gqa_rearrange_vt(const float* __restrict__ in, __bf16* __restrict__ out, int total4) {
  const int t = blockIdx.x * blockDim.x + threadIdx.x;
  if (t >= total4) return;
  const size_t i4 = (size_t)t * 4;
  const int d  = (int)(i4 & 127);
  const int kv = (int)((i4 >> 7) & 3);
  const int s  = (int)((i4 >> 9) & 2047);
  const int b  = (int)(i4 >> 20);
  float4 v = *reinterpret_cast<const float4*>(in + i4);
  __bf16* o = out + ((size_t)(b * NKV + kv) * HD + d) * SEQ + s;
  o[0 * SEQ] = (__bf16)v.x;
  o[1 * SEQ] = (__bf16)v.y;
  o[2 * SEQ] = (__bf16)v.z;
  o[3 * SEQ] = (__bf16)v.w;
}

// ---------------------------------------------------------------------------
// Flash attention: grid (S/64 q-blocks, B*NH). 128 threads = 4 waves, each
// wave owns 16 query rows. K/V^T streamed in 64-key blocks through
// double-buffered LDS with async global->LDS staging.
// ---------------------------------------------------------------------------
__global__ __launch_bounds__(128) void gqa_flash_attn(const __bf16* __restrict__ Q,
                                                      const __bf16* __restrict__ Km,
                                                      const __bf16* __restrict__ Vt,
                                                      __bf16* __restrict__ Out) {
  __shared__ __bf16 sQ[64 * 128];          // [qrow][d]
  __shared__ __bf16 sK[2][64 * 128];       // [krow][d]  (B-operand layout for Q*K^T)
  __shared__ __bf16 sVt[2][128 * 64];      // [d][krow]  (B-operand layout for P*V)
  __shared__ __bf16 sP[4][16 * 64];        // per-wave P strip, [row][kcol]

  const int tid  = threadIdx.x;
  const int lane = tid & 31, wave = tid >> 5;
  const int lg = lane >> 4, ll = lane & 15;
  const int bh = blockIdx.y;
  const int b = bh >> 4, h = bh & 15;
  const int kvh = h >> 2;

  const __bf16* qbase  = Q  + ((size_t)bh * SEQ + blockIdx.x * 64) * HD;
  const __bf16* kbase  = Km + (size_t)(b * NKV + kvh) * SEQ * HD;
  const __bf16* vtbase = Vt + (size_t)(b * NKV + kvh) * HD * SEQ;

  // stage Q block [64][128] (async)
  for (int c = tid; c < 1024; c += 128) {
    const int r = c >> 4, c8 = (c & 15) * 8;
    async_copy16(sQ + r * 128 + c8, qbase + r * 128 + c8);
  }
  // stage K/V block 0
  for (int c = tid; c < 1024; c += 128) {
    const int r = c >> 4, c8 = (c & 15) * 8;
    async_copy16(sK[0] + r * 128 + c8, kbase + r * 128 + c8);
    const int d = c >> 3, s8 = (c & 7) * 8;
    async_copy16(sVt[0] + d * 64 + s8, vtbase + (size_t)d * SEQ + s8);
  }
  wait_async0();
  __syncthreads();

  const v8f vzero = {0.f, 0.f, 0.f, 0.f, 0.f, 0.f, 0.f, 0.f};
  v8f O[8];
  float mrow[8], lrow[8];
#pragma unroll
  for (int j = 0; j < 8; ++j) O[j] = vzero;
#pragma unroll
  for (int r = 0; r < 8; ++r) { mrow[r] = -1e30f; lrow[r] = 0.f; }

  const float scale = 0.08838834764831845f; // 1/sqrt(128)

  for (int kb = 0; kb < SEQ / 64; ++kb) {
    const int cur = kb & 1;
    if (kb + 1 < SEQ / 64) {             // stage next K/V^T block while computing
      const int nxt = cur ^ 1;
      const __bf16* kp = kbase + (size_t)(kb + 1) * 64 * HD;
      for (int c = tid; c < 1024; c += 128) {
        const int r = c >> 4, c8 = (c & 15) * 8;
        async_copy16(sK[nxt] + r * 128 + c8, kp + r * 128 + c8);
        const int d = c >> 3, s8 = (c & 7) * 8;
        async_copy16(sVt[nxt] + d * 64 + s8, vtbase + (size_t)d * SEQ + (kb + 1) * 64 + s8);
      }
    }

    // ---- scores: 16x64 per wave = 4 n-tiles x 4 k-steps of WMMA ----
    v8f acc[4];
#pragma unroll
    for (int j = 0; j < 4; ++j) acc[j] = vzero;
#pragma unroll
    for (int t = 0; t < 4; ++t) {
      const __bf16* ap = sQ + (wave * 16 + ll) * 128 + t * 32 + 16 * lg;
      v16bf af = ld_frag2(ap, ap + 8);
#pragma unroll
      for (int j = 0; j < 4; ++j) {
        const __bf16* bp = sK[cur] + (j * 16 + ll) * 128 + t * 32 + 16 * lg;
        acc[j] = wmma_bf16f32(af, ld_frag2(bp, bp + 8), acc[j]);
      }
    }

    // ---- online softmax (row = r + 8*lg for this lane) ----
#pragma unroll
    for (int r = 0; r < 8; ++r) {
      float mx = fmaxf(fmaxf(acc[0][r], acc[1][r]), fmaxf(acc[2][r], acc[3][r]));
#pragma unroll
      for (int mk = 8; mk >= 1; mk >>= 1) mx = fmaxf(mx, __shfl_xor(mx, mk, 32));
      mx *= scale;
      const float mnew = fmaxf(mrow[r], mx);
      const float cf = __expf(mrow[r] - mnew);
      mrow[r] = mnew;
      float rs = 0.f;
#pragma unroll
      for (int j = 0; j < 4; ++j) {
        const float p = __expf(acc[j][r] * scale - mnew);
        acc[j][r] = p;
        rs += p;
      }
#pragma unroll
      for (int mk = 8; mk >= 1; mk >>= 1) rs += __shfl_xor(rs, mk, 32);
      lrow[r] = lrow[r] * cf + rs;
#pragma unroll
      for (int j = 0; j < 8; ++j) O[j][r] *= cf;
    }

    // ---- write P (bf16) to wave-private LDS strip, re-read as A fragments ----
    __bf16* ps = sP[wave];
#pragma unroll
    for (int j = 0; j < 4; ++j)
#pragma unroll
      for (int r = 0; r < 8; ++r)
        ps[(r + 8 * lg) * 64 + j * 16 + ll] = (__bf16)acc[j][r];

    // ---- P x V: 8 headdim n-tiles x 2 k-steps over 64 keys ----
#pragma unroll
    for (int t = 0; t < 2; ++t) {
      const __bf16* ap = ps + ll * 64 + t * 32 + 16 * lg;
      v16bf af = ld_frag2(ap, ap + 8);
#pragma unroll
      for (int j = 0; j < 8; ++j) {
        const __bf16* bp = sVt[cur] + (j * 16 + ll) * 64 + t * 32 + 16 * lg;
        O[j] = wmma_bf16f32(af, ld_frag2(bp, bp + 8), O[j]);
      }
    }
    wait_async0();
    __syncthreads();
  }

  // ---- epilogue: out token-major [B*S][DIM], col = h*128 + d ----
#pragma unroll
  for (int j = 0; j < 8; ++j)
#pragma unroll
    for (int r = 0; r < 8; ++r) {
      const float val = O[j][r] / lrow[r];
      const size_t row = (size_t)b * SEQ + blockIdx.x * 64 + wave * 16 + r + 8 * lg;
      Out[row * DIMN + h * HD + j * 16 + ll] = (__bf16)val;
    }
}

// ---------------------------------------------------------------------------
// Launch
// ---------------------------------------------------------------------------
extern "C" void kernel_launch(void* const* d_in, const int* in_sizes, int n_in,
                              void* d_out, int out_size, void* d_ws, size_t ws_size,
                              hipStream_t stream) {
  (void)in_sizes; (void)n_in; (void)out_size; (void)ws_size;
  const float* x   = (const float*)d_in[0];
  const float* Wq  = (const float*)d_in[1];
  const float* Wk  = (const float*)d_in[2];
  const float* Wv  = (const float*)d_in[3];
  const float* Wo  = (const float*)d_in[4];
  const float* qnw = (const float*)d_in[5];
  const float* knw = (const float*)d_in[6];
  float* out = (float*)d_out;

  char* ws = (char*)d_ws;
  size_t off = 0;
  __bf16* x_bf  = (__bf16*)(ws + off); off += (size_t)MTOT * DIMN * 2;
  __bf16* wq_t  = (__bf16*)(ws + off); off += (size_t)DIMN * DIMN * 2;  // [N][K]
  __bf16* wk_t  = (__bf16*)(ws + off); off += (size_t)DIMN * DKV * 2;
  __bf16* wv_t  = (__bf16*)(ws + off); off += (size_t)DIMN * DKV * 2;
  __bf16* wo_t  = (__bf16*)(ws + off); off += (size_t)DIMN * DIMN * 2;
  float*  q_raw = (float*)(ws + off);  off += (size_t)MTOT * DIMN * 4;
  float*  k_raw = (float*)(ws + off);  off += (size_t)MTOT * DKV * 4;
  float*  v_raw = (float*)(ws + off);  off += (size_t)MTOT * DKV * 4;
  __bf16* q_bf  = (__bf16*)(ws + off); off += (size_t)MTOT * DIMN * 2;
  __bf16* k_bf  = (__bf16*)(ws + off); off += (size_t)MTOT * DKV * 2;
  __bf16* vt_bf = (__bf16*)(ws + off); off += (size_t)MTOT * DKV * 2;
  __bf16* a_bf  = (__bf16*)(ws + off); off += (size_t)MTOT * DIMN * 2;

  const int tpb = 256;
  // 1) convert x; convert+transpose weights
  gqa_f32_to_bf16<<<(MTOT * DIMN / 4) / tpb, tpb, 0, stream>>>(x, x_bf, MTOT * DIMN / 4);
  gqa_f32_to_bf16_t<<<(DIMN * DIMN / 4) / tpb, tpb, 0, stream>>>(Wq, wq_t, DIMN, DIMN, DIMN * DIMN / 4);
  gqa_f32_to_bf16_t<<<(DIMN * DKV / 4) / tpb, tpb, 0, stream>>>(Wk, wk_t, DKV, DIMN, DIMN * DKV / 4);
  gqa_f32_to_bf16_t<<<(DIMN * DKV / 4) / tpb, tpb, 0, stream>>>(Wv, wv_t, DKV, DIMN, DIMN * DKV / 4);
  gqa_f32_to_bf16_t<<<(DIMN * DIMN / 4) / tpb, tpb, 0, stream>>>(Wo, wo_t, DIMN, DIMN, DIMN * DIMN / 4);

  // 2) QKV projections (WMMA GEMMs, async double-buffered)
  gqa_gemm_bf16f32<<<dim3(MTOT / 128, DIMN / 128), 256, 0, stream>>>(x_bf, wq_t, q_raw, DIMN, DIMN);
  gqa_gemm_bf16f32<<<dim3(MTOT / 128, DKV / 128),  256, 0, stream>>>(x_bf, wk_t, k_raw, DKV,  DIMN);
  gqa_gemm_bf16f32<<<dim3(MTOT / 128, DKV / 128),  256, 0, stream>>>(x_bf, wv_t, v_raw, DKV,  DIMN);

  // 3) RMSNorm + RoPE (q, k) and V transpose/rearrange
  gqa_rmsnorm_rope<<<(MTOT * NH) / 4,  128, 0, stream>>>(q_raw, qnw, q_bf, NH,  MTOT * NH);
  gqa_rmsnorm_rope<<<(MTOT * NKV) / 4, 128, 0, stream>>>(k_raw, knw, k_bf, NKV, MTOT * NKV);
  gqa_rearrange_vt<<<(MTOT * DKV / 4) / tpb, tpb, 0, stream>>>(v_raw, vt_bf, MTOT * DKV / 4);

  // 4) flash attention (WMMA, async double-buffered K/V^T)
  gqa_flash_attn<<<dim3(SEQ / 64, BATCH * NH), 128, 0, stream>>>(q_bf, k_bf, vt_bf, a_bf);

  // 5) output projection (WMMA GEMM, f32 out)
  gqa_gemm_bf16f32<<<dim3(MTOT / 128, DIMN / 128), 256, 0, stream>>>(a_bf, wo_t, out, DIMN, DIMN);
}
